// StatefulCausalDiffAttentionHead_34342558498820
// MI455X (gfx1250) — compile-verified, matrix operand order checked by reading, not run on hardware
//
#include <hip/hip_runtime.h>
#include <hip/hip_bf16.h>

// ---------------------------------------------------------------------------
// StatefulCausalDiffAttentionHead for MI455X (gfx1250, wave32, WMMA bf16)
// ---------------------------------------------------------------------------
// Pipeline (all on `stream`, workspace in d_ws):
//   k0: convert 9 fp32 weight matrices (1024x128) -> bf16 transposed (128x1024)
//   k1: async-stage x tile to LDS (global_load_async_to_lds_b128, inline asm),
//       per-segment LayerNorm, Q/K/V projections via v_wmma_f32_16x16x32_bf16
//       (V stored TRANSPOSED: [dim][token])
//   k2: lambda scalar
//   k3: one-pass (flash) dual-softmax causal attention + diff + output LN
//
// WMMA register layouts assumed (CDNA5 ISA 7.12.2):
//   A (16x32 bf16): lane = row (lane&15), half-wave selects K 0-7/16-23 vs
//                   8-15/24-31, VGPR v holds packed pair {kbase, kbase+1}
//   B (32x16 bf16): symmetric, lane = column
//   C/D (16x16 f32): lane = column, VGPR v = row v (+8 for upper half-wave)
//
// Async-LDS note: per ISA aperture rules (addr[31:0] = LDS offset), the low
// 32 bits of a generic pointer into __shared__ ARE the LDS byte offset, so
// (unsigned)(size_t)&shmem[i] is a valid VDST for global_load_async_to_lds_*.
// ---------------------------------------------------------------------------

typedef __attribute__((ext_vector_type(16))) __bf16 v16bf;
typedef __attribute__((ext_vector_type(8)))  float  v8f;

#define B_          8
#define T_          2304
#define STATE_      128
#define DIN         1024
#define D2          128           // 2*DIM_K == 2*DIM_V
#define NT_TILES    144           // T_/16
#define SCALE       0.125f        // 1/sqrt(64)
#define LOG2E       1.4426950408889634f
#define NEGINF      (-1e30f)
#define LAMBDA_INIT 0.6192834728526787f   // 0.8 - 0.6*exp(-1.2)

#define USE_ASYNC_LDS_ASM 1

union BFPack {
    v16bf        v;
    unsigned int u[8];
    __bf16       h[16];
};

__device__ __forceinline__ int kbase(int v, int half) {
    return (v < 4 ? 2 * v : 16 + 2 * (v - 4)) + half * 8;
}

__device__ __forceinline__ v8f wmma_bf16(v16bf a, v16bf b, v8f c) {
    return __builtin_amdgcn_wmma_f32_16x16x32_bf16(
        /*neg_a=*/false, a, /*neg_b=*/false, b,
        /*c_mod=*/(short)0, c, /*reuse_a=*/false, /*reuse_b=*/false);
}

__device__ __forceinline__ float grpmax16(float x) {
    #pragma unroll
    for (int m = 1; m < 16; m <<= 1) x = fmaxf(x, __shfl_xor(x, m));
    return x;    // xor masks < 16 stay within each 16-lane half-wave group
}
__device__ __forceinline__ float grpsum16(float x) {
    #pragma unroll
    for (int m = 1; m < 16; m <<= 1) x += __shfl_xor(x, m);
    return x;
}

// --------------------------- k0: weight convert -----------------------------
__global__ __launch_bounds__(256) void wcvt_kernel(
    const float* __restrict__ w0, const float* __restrict__ w1,
    const float* __restrict__ w2, const float* __restrict__ w3,
    const float* __restrict__ w4, const float* __restrict__ w5,
    const float* __restrict__ w6, const float* __restrict__ w7,
    const float* __restrict__ w8, __bf16* __restrict__ wt)
{
    const int per = DIN * D2;                       // 131072
    int i = blockIdx.x * blockDim.x + threadIdx.x;
    if (i >= 9 * per) return;
    const float* ws[9] = {w0, w1, w2, w3, w4, w5, w6, w7, w8};
    int mm = i / per, r = i % per;
    int k = r / D2, n = r % D2;
    // transposed store: wt[m][n*1024 + k]  (so B-tile pairs are contiguous)
    wt[(size_t)mm * per + (size_t)n * DIN + k] = (__bf16)ws[mm][r];
}

// -------------------- k1: segment LayerNorm + projections -------------------
__global__ __launch_bounds__(128) void proj_kernel(
    const float* __restrict__ x,
    const float* __restrict__ g_in, const float* __restrict__ b_in,
    const float* __restrict__ g_ss, const float* __restrict__ b_ss,
    const float* __restrict__ g_se, const float* __restrict__ b_se,
    const __bf16* __restrict__ wt,
    __bf16* __restrict__ Q, __bf16* __restrict__ K, __bf16* __restrict__ Vt)
{
    __shared__ __align__(16) float  xf[16 * DIN];   // 64 KB staged fp32 tile
    __shared__ __align__(16) __bf16 xa[16 * DIN];   // 32 KB normalized bf16

    int bid  = blockIdx.x;
    int b    = bid / NT_TILES;
    int t0   = (bid % NT_TILES) * 16;
    int tid  = threadIdx.x;
    int wave = tid >> 5;
    int lane = tid & 31;

    // 16-aligned tiles never straddle segments (STATE=128 is a multiple of 16)
    int seg = (t0 < STATE_) ? 1 : ((t0 >= T_ - STATE_) ? 2 : 0);
    const float* g  = (seg == 0) ? g_in : (seg == 1) ? g_ss : g_se;
    const float* bb = (seg == 0) ? b_in : (seg == 1) ? b_ss : b_se;
    int wslotBase   = (seg == 0) ? 0 : (seg == 1) ? 3 : 6;

    // ---- stage the contiguous 16x1024 fp32 tile into LDS -------------------
    const float* xbase = x + ((size_t)b * T_ + t0) * DIN;   // 16384 floats
#if USE_ASYNC_LDS_ASM
    #pragma unroll
    for (int it = 0; it < 32; ++it) {               // 128 thr * 32 * 16B = 64KB
        int i4 = (tid + it * 128) * 4;
        unsigned ldsoff = (unsigned)(size_t)&xf[i4];    // low 32b = LDS offset
        asm volatile("global_load_async_to_lds_b128 %0, %1, off"
                     :
                     : "v"(ldsoff), "v"(xbase + i4)
                     : "memory");
    }
    asm volatile("s_wait_asynccnt 0x0" ::: "memory");
#else
    {
        const float4* src = (const float4*)xbase;
        float4*       dst = (float4*)xf;
        #pragma unroll
        for (int it = 0; it < 32; ++it) dst[tid + it * 128] = src[tid + it * 128];
    }
#endif
    __syncthreads();

    // ---- LayerNorm 16 rows (4 waves x 4 rows) from LDS, bf16 back to LDS ---
    for (int r = wave; r < 16; r += 4) {
        float s = 0.f, ss = 0.f;
        #pragma unroll
        for (int j = 0; j < DIN / 32; ++j) {
            float vx = xf[r * DIN + lane + 32 * j];
            s += vx; ss += vx * vx;
        }
        #pragma unroll
        for (int mk = 1; mk < 32; mk <<= 1) {
            s  += __shfl_xor(s,  mk);
            ss += __shfl_xor(ss, mk);
        }
        float mean = s * (1.f / DIN);
        float var  = ss * (1.f / DIN) - mean * mean;
        float rstd = rsqrtf(var + 1e-5f);
        #pragma unroll
        for (int j = 0; j < DIN / 32; ++j) {
            int c = lane + 32 * j;
            xa[r * DIN + c] = (__bf16)((xf[r * DIN + c] - mean) * rstd * g[c] + bb[c]);
        }
    }
    __syncthreads();

    int m = lane & 15, half = lane >> 4;

    // ---- 24 output tiles (q:8, k:8, v:8), 6 per wave, K-loop of 32 ---------
    #pragma unroll
    for (int i = 0; i < 6; ++i) {
        int idx = wave * 6 + i;
        int mat = idx >> 3;                          // 0=q 1=k 2=v
        int nt  = idx & 7;
        const __bf16* wmat = wt + (size_t)(wslotBase + mat) * (DIN * D2);
        int ncol = nt * 16 + m;

        v8f acc = {0.f, 0.f, 0.f, 0.f, 0.f, 0.f, 0.f, 0.f};
        for (int kk = 0; kk < DIN; kk += 32) {
            BFPack a, bm;
            #pragma unroll
            for (int v = 0; v < 8; ++v) {
                int kb = kbase(v, half);
                a.u[v]  = *(const unsigned int*)&xa[m * DIN + kk + kb];
                bm.u[v] = *(const unsigned int*)&wmat[(size_t)ncol * DIN + kk + kb];
            }
            acc = wmma_bf16(a.v, bm.v, acc);
        }
        if (mat == 2) {
            // V stored transposed: Vt[b][dim][token] so attention B-tiles
            // (32 tokens x 16 dims) become contiguous dword pair loads
            __bf16* dstv = Vt + (size_t)b * T_ * D2 + (size_t)ncol * T_;
            #pragma unroll
            for (int v = 0; v < 8; ++v)
                dstv[t0 + v + half * 8] = (__bf16)acc[v];
        } else {
            __bf16* dst = (mat == 0 ? Q : K) + ((size_t)b * T_ + t0) * D2;
            #pragma unroll
            for (int v = 0; v < 8; ++v)
                dst[(v + half * 8) * D2 + ncol] = (__bf16)acc[v];
        }
    }
}

// ------------------------------ k2: lambda ----------------------------------
__global__ __launch_bounds__(32) void lambda_kernel(
    const float* __restrict__ lq1, const float* __restrict__ lq2,
    const float* __restrict__ lk1, const float* __restrict__ lk2,
    float* __restrict__ lam)
{
    int lane = threadIdx.x;
    float d1 = lq1[lane] * lk1[lane] + lq1[lane + 32] * lk1[lane + 32];
    float d2 = lq2[lane] * lk2[lane] + lq2[lane + 32] * lk2[lane + 32];
    #pragma unroll
    for (int mk = 1; mk < 32; mk <<= 1) {
        d1 += __shfl_xor(d1, mk);
        d2 += __shfl_xor(d2, mk);
    }
    if (lane == 0) *lam = expf(d1) - expf(d2) + LAMBDA_INIT;
}

// -------------------- k3: flash dual-softmax diff attention -----------------
__global__ __launch_bounds__(32) void attn_kernel(
    const __bf16* __restrict__ Q, const __bf16* __restrict__ K,
    const __bf16* __restrict__ Vt, const float* __restrict__ lamPtr,
    const float* __restrict__ g_out, const float* __restrict__ b_out,
    float* __restrict__ out)
{
    __shared__ __align__(16) __bf16 pbuf[2][16 * 32];   // C->A transpose bounce

    int bid  = blockIdx.x;
    int b    = bid / NT_TILES;
    int q0   = (bid % NT_TILES) * 16;
    int lane = threadIdx.x & 31;
    int m    = lane & 15, half = lane >> 4;

    const __bf16* Qb  = Q  + (size_t)b * T_ * D2;
    const __bf16* Kb  = K  + (size_t)b * T_ * D2;
    const __bf16* Vtb = Vt + (size_t)b * T_ * D2;        // [dim][token]

    // Q1/Q2 as A-tiles (16 x 64 each = 2 k-subtiles), contiguous dword loads
    BFPack q1a[2], q2a[2];
    #pragma unroll
    for (int t = 0; t < 2; ++t) {
        #pragma unroll
        for (int v = 0; v < 8; ++v) {
            int kb = t * 32 + kbase(v, half);
            q1a[t].u[v] = *(const unsigned int*)&Qb[(size_t)(q0 + m) * D2 + kb];
            q2a[t].u[v] = *(const unsigned int*)&Qb[(size_t)(q0 + m) * D2 + 64 + kb];
        }
    }

    v8f o1acc[8], o2acc[8];
    #pragma unroll
    for (int n = 0; n < 8; ++n) {
        o1acc[n] = {0.f, 0.f, 0.f, 0.f, 0.f, 0.f, 0.f, 0.f};
        o2acc[n] = {0.f, 0.f, 0.f, 0.f, 0.f, 0.f, 0.f, 0.f};
    }
    float m1v[8], l1v[8], m2v[8], l2v[8];
    #pragma unroll
    for (int v = 0; v < 8; ++v) { m1v[v] = -3e38f; m2v[v] = -3e38f; l1v[v] = 0.f; l2v[v] = 0.f; }

    for (int j0 = 0; j0 < q0 + 16; j0 += 32) {      // 32 keys / step (causal)
        if (j0 + 32 < q0 + 16) {                    // prefetch next K/V tile
            __builtin_prefetch(&Kb[(size_t)(j0 + 32) * D2], 0, 1);
            __builtin_prefetch(&Vtb[(size_t)(lane * 4) * T_ + j0 + 32], 0, 1);
        }

        // ---- S1/S2 = Q{1,2} . K{1,2}^T : Kt B-tiles are contiguous dwords --
        v8f s1[2], s2[2];
        #pragma unroll
        for (int nt = 0; nt < 2; ++nt) {
            s1[nt] = {0.f, 0.f, 0.f, 0.f, 0.f, 0.f, 0.f, 0.f};
            s2[nt] = {0.f, 0.f, 0.f, 0.f, 0.f, 0.f, 0.f, 0.f};
            int tok = j0 + nt * 16 + m;
            #pragma unroll
            for (int t = 0; t < 2; ++t) {
                BFPack bk1, bk2;
                #pragma unroll
                for (int v = 0; v < 8; ++v) {
                    int kb = t * 32 + kbase(v, half);
                    bk1.u[v] = *(const unsigned int*)&Kb[(size_t)tok * D2 + kb];
                    bk2.u[v] = *(const unsigned int*)&Kb[(size_t)tok * D2 + 64 + kb];
                }
                s1[nt] = wmma_bf16(q1a[t].v, bk1.v, s1[nt]);
                s2[nt] = wmma_bf16(q2a[t].v, bk2.v, s2[nt]);
            }
        }

        // ---- scale + causal mask (C layout: lane=col, vgpr=row) ------------
        bool needMask = (j0 + 31 > q0);
        #pragma unroll
        for (int nt = 0; nt < 2; ++nt) {
            #pragma unroll
            for (int v = 0; v < 8; ++v) {
                float sv1 = s1[nt][v] * SCALE;
                float sv2 = s2[nt][v] * SCALE;
                if (needMask) {
                    int row = q0 + v + half * 8;
                    int col = j0 + nt * 16 + m;
                    if (col > row) { sv1 = NEGINF; sv2 = NEGINF; }
                }
                s1[nt][v] = sv1;
                s2[nt][v] = sv2;
            }
        }

        // ---- online softmax stats; turn S into P in-place ------------------
        float a1v[8], a2v[8];
        #pragma unroll
        for (int v = 0; v < 8; ++v) {
            float t1  = grpmax16(fmaxf(s1[0][v], s1[1][v]));
            float mn1 = fmaxf(m1v[v], t1);
            float al1 = exp2f((m1v[v] - mn1) * LOG2E);
            float p10 = exp2f((s1[0][v] - mn1) * LOG2E);
            float p11 = exp2f((s1[1][v] - mn1) * LOG2E);
            l1v[v] = l1v[v] * al1 + grpsum16(p10 + p11);
            m1v[v] = mn1; a1v[v] = al1;
            s1[0][v] = p10; s1[1][v] = p11;

            float t2  = grpmax16(fmaxf(s2[0][v], s2[1][v]));
            float mn2 = fmaxf(m2v[v], t2);
            float al2 = exp2f((m2v[v] - mn2) * LOG2E);
            float p20 = exp2f((s2[0][v] - mn2) * LOG2E);
            float p21 = exp2f((s2[1][v] - mn2) * LOG2E);
            l2v[v] = l2v[v] * al2 + grpsum16(p20 + p21);
            m2v[v] = mn2; a2v[v] = al2;
            s2[0][v] = p20; s2[1][v] = p21;
        }
        v8f av1, av2;
        #pragma unroll
        for (int v = 0; v < 8; ++v) { av1[v] = a1v[v]; av2[v] = a2v[v]; }
        #pragma unroll
        for (int n = 0; n < 8; ++n) { o1acc[n] *= av1; o2acc[n] *= av2; }

        // ---- P (C layout) -> LDS -> A layout -------------------------------
        #pragma unroll
        for (int nt = 0; nt < 2; ++nt) {
            #pragma unroll
            for (int v = 0; v < 8; ++v) {
                int row = v + half * 8, col = nt * 16 + m;
                pbuf[0][row * 32 + col] = (__bf16)s1[nt][v];
                pbuf[1][row * 32 + col] = (__bf16)s2[nt][v];
            }
        }
        __syncthreads();                            // single wave: ~S_NOP cost
        BFPack pa1, pa2;
        #pragma unroll
        for (int v = 0; v < 8; ++v) {
            int kb = kbase(v, half);
            pa1.u[v] = *(const unsigned int*)&pbuf[0][m * 32 + kb];
            pa2.u[v] = *(const unsigned int*)&pbuf[1][m * 32 + kb];
        }
        __syncthreads();

        // ---- O{1,2} += P{1,2} @ V : Vt B-tiles are contiguous dwords -------
        #pragma unroll
        for (int n = 0; n < 8; ++n) {
            BFPack bv;
            int col = n * 16 + m;
            #pragma unroll
            for (int v = 0; v < 8; ++v) {
                int k = kbase(v, half);
                bv.u[v] = *(const unsigned int*)&Vtb[(size_t)col * T_ + j0 + k];
            }
            o1acc[n] = wmma_bf16(pa1.v, bv.v, o1acc[n]);
            o2acc[n] = wmma_bf16(pa2.v, bv.v, o2acc[n]);
        }
    }

    // ---- finalize: diff, output LayerNorm, store fp32 ----------------------
    float lam = *lamPtr;
    float att[8][8];
    #pragma unroll
    for (int v = 0; v < 8; ++v) {
        float inv1 = 1.f / l1v[v], inv2 = 1.f / l2v[v];
        #pragma unroll
        for (int n = 0; n < 8; ++n)
            att[v][n] = o1acc[n][v] * inv1 - lam * (o2acc[n][v] * inv2);
    }
    #pragma unroll
    for (int v = 0; v < 8; ++v) {
        float s = 0.f;
        #pragma unroll
        for (int n = 0; n < 8; ++n) s += att[v][n];
        float mean = grpsum16(s) * (1.f / 128.f);
        float ss = 0.f;
        #pragma unroll
        for (int n = 0; n < 8; ++n) { float d = att[v][n] - mean; ss += d * d; }
        float rstd = rsqrtf(grpsum16(ss) * (1.f / 128.f) + 1e-5f);
        int row = q0 + v + half * 8;
        #pragma unroll
        for (int n = 0; n < 8; ++n) {
            int col = n * 16 + m;
            out[((size_t)b * T_ + row) * D2 + col] =
                (att[v][n] - mean) * rstd * g_out[col] + b_out[col];
        }
    }
}

// ---------------------------------------------------------------------------
extern "C" void kernel_launch(void* const* d_in, const int* in_sizes, int n_in,
                              void* d_out, int out_size, void* d_ws, size_t ws_size,
                              hipStream_t stream) {
    const float* x = (const float*)d_in[0];
    const float* W[9];
    for (int i = 0; i < 9; ++i) W[i] = (const float*)d_in[1 + i];
    const float* g_in  = (const float*)d_in[10];
    const float* b_in  = (const float*)d_in[11];
    const float* g_ss  = (const float*)d_in[12];
    const float* b_ss  = (const float*)d_in[13];
    const float* g_se  = (const float*)d_in[14];
    const float* b_se  = (const float*)d_in[15];
    const float* g_out = (const float*)d_in[16];
    const float* b_out = (const float*)d_in[17];
    const float* lq1   = (const float*)d_in[18];
    const float* lq2   = (const float*)d_in[19];
    const float* lk1   = (const float*)d_in[20];
    const float* lk2   = (const float*)d_in[21];

    char*  ws  = (char*)d_ws;
    size_t off = 0;
    __bf16* wt = (__bf16*)(ws + off); off += (size_t)9 * DIN * D2 * sizeof(__bf16);
    __bf16* Qw = (__bf16*)(ws + off); off += (size_t)B_ * T_ * D2 * sizeof(__bf16);
    __bf16* Kw = (__bf16*)(ws + off); off += (size_t)B_ * T_ * D2 * sizeof(__bf16);
    __bf16* Vw = (__bf16*)(ws + off); off += (size_t)B_ * T_ * D2 * sizeof(__bf16);
    float*  lam = (float*)(ws + off);

    int nW = 9 * DIN * D2;
    wcvt_kernel<<<(nW + 255) / 256, 256, 0, stream>>>(
        W[0], W[1], W[2], W[3], W[4], W[5], W[6], W[7], W[8], wt);

    proj_kernel<<<B_ * NT_TILES, 128, 0, stream>>>(
        x, g_in, b_in, g_ss, b_ss, g_se, b_se, wt, Qw, Kw, Vw);

    lambda_kernel<<<1, 32, 0, stream>>>(lq1, lq2, lk1, lk2, lam);

    attn_kernel<<<B_ * NT_TILES, 32, 0, stream>>>(
        Qw, Kw, Vw, lam, g_out, b_out, (float*)d_out);
}
